// ProposalBranch_conf_54528904790789
// MI455X (gfx1250) — compile-verified, matrix operand order checked by reading, not run on hardware
//
#include <hip/hip_runtime.h>

#define PCc   512
#define CINc  512
#define BBc   2
#define TTc   128
#define T2c   512
#define GRPS  32
#define EPSc  1e-5f
#define NTILES 8            // N = 128 -> 8 tiles of 16 for every GEMM here

typedef __attribute__((ext_vector_type(16))) __bf16 v16bf;
typedef __attribute__((ext_vector_type(8)))  float  v8f;

union Frag { v16bf v; unsigned int u[8]; };

__device__ __forceinline__ unsigned int pack_bf2(float a, float b) {
  unsigned int ua = __float_as_uint(a);
  unsigned int ub = __float_as_uint(b);
  ua = (ua + 0x7FFFu + ((ua >> 16) & 1u)) >> 16;   // round-to-nearest-even bf16
  ub = (ub + 0x7FFFu + ((ub >> 16) & 1u)) >> 16;
  return (ua & 0xFFFFu) | (ub << 16);
}

// K offset within a 32-wide block for fragment dword i (i in 0..7), half = lane>>4
__device__ __forceinline__ int frag_koff(int i, int half) {
  return ((i & 4) << 2) + (half << 3) + ((i & 3) << 1);
}

// ---------------------------------------------------------------------------
// Pack weights W[M,K] fp32 -> A-fragment layout Wp[mt][kblk][lane][i] (bf16x2)
// ---------------------------------------------------------------------------
__global__ void k_pack_a(const float* __restrict__ W, unsigned int* __restrict__ Wp,
                         int K, int total) {
  const int id = blockIdx.x * blockDim.x + threadIdx.x;
  if (id >= total) return;
  const int i    = id & 7;
  const int lane = (id >> 3) & 31;
  const int rest = id >> 8;
  const int KB   = K >> 5;
  const int kblk = rest % KB;
  const int mt   = rest / KB;
  const int row  = mt * 16 + (lane & 15);
  const int kk   = kblk * 32 + frag_koff(i, lane >> 4);
  const float2 w = *(const float2*)(W + (size_t)row * K + kk);
  Wp[id] = pack_bf2(w.x, w.y);
}

// ---------------------------------------------------------------------------
// Pack activations X[B,K,N=128] fp32 -> B-fragment layout Xp[b][kblk][nt][lane][i]
// ---------------------------------------------------------------------------
__global__ void k_pack_b(const float* __restrict__ X, unsigned int* __restrict__ Xp,
                         int K, int total) {
  const int id = blockIdx.x * blockDim.x + threadIdx.x;
  if (id >= total) return;
  const int i    = id & 7;
  const int lane = (id >> 3) & 31;
  const int rest = id >> 8;
  const int KB   = K >> 5;
  const int nt   = rest % NTILES;
  const int kblk = (rest / NTILES) % KB;
  const int b    = rest / (NTILES * KB);
  const int n    = nt * 16 + (lane & 15);
  const int kk   = kblk * 32 + frag_koff(i, lane >> 4);
  const float x0 = X[((size_t)b * K + kk) * TTc + n];
  const float x1 = X[((size_t)b * K + kk + 1) * TTc + n];
  Xp[id] = pack_bf2(x0, x1);
}

// ---------------------------------------------------------------------------
// Anchor-gather fused with packing: emits gathered matrix [B, C*KK, 128]
// directly in packed-bf16 B-fragment layout. (kk even -> both elements share c)
// ---------------------------------------------------------------------------
__global__ void k_gather_pack(const float* __restrict__ feat, const int* __restrict__ idx,
                              unsigned int* __restrict__ Gp, int K, int kklog, int Tf,
                              int C, int total) {
  const int id = blockIdx.x * blockDim.x + threadIdx.x;
  if (id >= total) return;
  const int i    = id & 7;
  const int lane = (id >> 3) & 31;
  const int rest = id >> 8;
  const int KB   = K >> 5;
  const int nt   = rest % NTILES;
  const int kblk = (rest / NTILES) % KB;
  const int b    = rest / (NTILES * KB);
  const int l    = nt * 16 + (lane & 15);
  const int kk   = kblk * 32 + frag_koff(i, lane >> 4);
  const int c = kk >> kklog;
  const int k = kk & ((1 << kklog) - 1);
  const int KK = 1 << kklog;
  const int t0 = idx[((size_t)b * TTc + l) * KK + k];
  const int t1 = idx[((size_t)b * TTc + l) * KK + k + 1];
  const float* fc = feat + ((size_t)b * C + c) * Tf;
  const float v0 = (t0 < 0) ? 0.f : fc[t0];
  const float v1 = (t1 < 0) ? 0.f : fc[t1];
  Gp[id] = pack_bf2(v0, v1);
}

// ---------------------------------------------------------------------------
// Packed-operand WMMA GEMM: Y[b, choff+m, n] = sum_k W[m,k]*X[b,k,n] + bias[m]
// grid: (N/64, M/128, B), block 256 = 8 waves; wave = 16M x 64N, 4 accumulators.
// B fragments staged through LDS (shared by all 8 waves); zero conversion VALU.
// ---------------------------------------------------------------------------
__global__ void k_gemm_pk(const unsigned int* __restrict__ Wp, const float* __restrict__ bias,
                          const unsigned int* __restrict__ Xp, float* __restrict__ Y,
                          int KB, long ldYb, int choff) {
  __shared__ __align__(16) unsigned int ldsb[1024];   // 4 ntiles x 32 lanes x 8 dwords
  const int tid  = threadIdx.x;
  const int wave = tid >> 5;
  const int lane = tid & 31;
  const int row  = lane & 15;
  const int half = lane >> 4;
  const int nb  = blockIdx.x * 64;
  const int nt0 = blockIdx.x * 4;
  const int mt  = blockIdx.y * 8 + wave;
  const int b   = blockIdx.z;

  const long ldXb = (long)KB * NTILES * 256;
  const unsigned int* Xb = Xp + (size_t)b * ldXb;
  const unsigned int* Wl = Wp + ((size_t)mt * KB) * 256 + lane * 8;

  v8f acc0 = {0.f,0.f,0.f,0.f,0.f,0.f,0.f,0.f};
  v8f acc1 = acc0, acc2 = acc0, acc3 = acc0;

  for (int kblk = 0; kblk < KB; ++kblk) {
    // stage this K-block's 4 B-fragments (1024 dwords) into LDS, coalesced b128
    const unsigned int* src = Xb + ((size_t)kblk * NTILES + nt0) * 256;
    const uint4 stg = *(const uint4*)(src + tid * 4);
    if (kblk + 1 < KB) {                               // global_prefetch_b8
      __builtin_prefetch(src + (size_t)NTILES * 256 + tid * 4, 0, 1);
      __builtin_prefetch(Wl + (size_t)(kblk + 1) * 256, 0, 1);
    }
    __syncthreads();
    *(uint4*)(ldsb + tid * 4) = stg;
    __syncthreads();

    // A fragment: 2 x b128, already in bf16 fragment order
    Frag fa;
    {
      const uint4 a0 = *(const uint4*)(Wl + (size_t)kblk * 256);
      const uint4 a1 = *(const uint4*)(Wl + (size_t)kblk * 256 + 4);
      fa.u[0] = a0.x; fa.u[1] = a0.y; fa.u[2] = a0.z; fa.u[3] = a0.w;
      fa.u[4] = a1.x; fa.u[5] = a1.y; fa.u[6] = a1.z; fa.u[7] = a1.w;
    }
#pragma unroll
    for (int j = 0; j < 4; ++j) {
      const uint4 b0 = *(const uint4*)(ldsb + j * 256 + lane * 8);
      const uint4 b1 = *(const uint4*)(ldsb + j * 256 + lane * 8 + 4);
      Frag fb;
      fb.u[0] = b0.x; fb.u[1] = b0.y; fb.u[2] = b0.z; fb.u[3] = b0.w;
      fb.u[4] = b1.x; fb.u[5] = b1.y; fb.u[6] = b1.z; fb.u[7] = b1.w;
      v8f* pa = (j == 0) ? &acc0 : (j == 1) ? &acc1 : (j == 2) ? &acc2 : &acc3;
      *pa = __builtin_amdgcn_wmma_f32_16x16x32_bf16(false, fa.v, false, fb.v,
                                                    (short)0, *pa, false, false);
    }
  }

  float* Yb = Y + (size_t)b * ldYb;
#pragma unroll
  for (int j = 0; j < 4; ++j) {
    const v8f acc = (j == 0) ? acc0 : (j == 1) ? acc1 : (j == 2) ? acc2 : acc3;
    const int n = nb + j * 16 + row;
#pragma unroll
    for (int r = 0; r < 8; ++r) {
      const int m = mt * 16 + r + (half << 3);
      Yb[(size_t)(choff + m) * TTc + n] = acc[r] + bias[m];
    }
  }
}

// ---------------------------------------------------------------------------
// GroupNorm statistics / apply, boundary max pool, index precompute
// ---------------------------------------------------------------------------
__global__ void k_gn_stats(const float* __restrict__ X, float* __restrict__ st,
                           int C, int Tl) {
  const int g = blockIdx.x, b = blockIdx.y;
  const int Cg = C / GRPS;
  const int n = Cg * Tl;
  const float* xb = X + ((size_t)b * C + (size_t)g * Cg) * Tl;
  float s = 0.f, s2 = 0.f;
  for (int i = threadIdx.x; i < n; i += blockDim.x) {
    const float v = xb[i];
    s += v; s2 += v * v;
  }
  __shared__ float sh[256], sh2[256];
  sh[threadIdx.x] = s; sh2[threadIdx.x] = s2;
  __syncthreads();
  for (int off = 128; off > 0; off >>= 1) {
    if ((int)threadIdx.x < off) {
      sh[threadIdx.x]  += sh[threadIdx.x + off];
      sh2[threadIdx.x] += sh2[threadIdx.x + off];
    }
    __syncthreads();
  }
  if (threadIdx.x == 0) {
    const float mean = sh[0] / (float)n;
    const float var  = sh2[0] / (float)n - mean * mean;
    st[(b * GRPS + g) * 2 + 0] = mean;
    st[(b * GRPS + g) * 2 + 1] = rsqrtf(var + EPSc);
  }
}

__global__ void k_gn_apply(const float* __restrict__ X, const float* __restrict__ st,
                           const float* __restrict__ gamma, const float* __restrict__ beta,
                           float* __restrict__ dst, int C, int Tl, int choff, int ldC,
                           int total) {
  const int i = blockIdx.x * blockDim.x + threadIdx.x;
  if (i >= total) return;
  const int t = i % Tl;
  const int c = (i / Tl) % C;
  const int b = i / (Tl * C);
  const int g = c / (C / GRPS);
  const float mean = st[(b * GRPS + g) * 2 + 0];
  const float rstd = st[(b * GRPS + g) * 2 + 1];
  const float v = (X[i] - mean) * rstd * gamma[c] + beta[c];
  dst[((size_t)b * ldC + choff + c) * Tl + t] = fmaxf(v, 0.f);
}

__global__ void k_bmax(const float* __restrict__ feat, const float* __restrict__ segs,
                       float* __restrict__ dst, int C, int Tf, int choff, int ldC,
                       int total) {
  const int i = blockIdx.x * blockDim.x + threadIdx.x;
  if (i >= total) return;
  const int n = i % TTc;
  const int c = (i / TTc) % C;
  const int b = i / (TTc * C);
  const int pr = (c < (C >> 1)) ? 0 : 2;
  const float s = segs[((size_t)b * TTc + n) * 4 + pr];
  const float e = segs[((size_t)b * TTc + n) * 4 + pr + 1];
  int t0 = (int)floorf(s); if (t0 < 0) t0 = 0;
  int t1 = (int)floorf(e); if (t1 > Tf - 1) t1 = Tf - 1;
  float m = 0.f;
  if (t0 <= t1) {
    m = -3.402823466e38f;
    const float* f = feat + ((size_t)b * C + c) * Tf;
    for (int t = t0; t <= t1; ++t) m = fmaxf(m, f[t]);
  }
  dst[((size_t)b * ldC + choff + c) * TTc + n] = m;
}

__global__ void k_mkidx(const float* __restrict__ anc, int* __restrict__ idx,
                        int Tf, int KK, int total) {
  const int i = blockIdx.x * blockDim.x + threadIdx.x;
  if (i >= total) return;
  const int k = i % KK;
  const int l = (i / KK) % TTc;
  const int b = i / (KK * TTc);
  const float a0 = anc[((size_t)b * TTc + l) * 2 + 0];
  const float a1 = anc[((size_t)b * TTc + l) * 2 + 1];
  const float left  = fminf(fmaxf(a0, 0.f), (float)Tf);
  const float right = fminf(fmaxf(a1, 0.f), (float)Tf);
  const bool valid = right > left;
  const int li = (int)floorf(left);
  const int ri = (int)floorf(right);
  int len = (ri + 1 < Tf ? ri + 1 : Tf) - li;
  if (len < 1) len = 1;
  int id = li + (k * len) / KK;
  if (id < 0) id = 0;
  if (id > Tf - 1) id = Tf - 1;
  idx[i] = valid ? id : -1;
}

extern "C" void kernel_launch(void* const* d_in, const int* in_sizes, int n_in,
                              void* d_out, int out_size, void* d_ws, size_t ws_size,
                              hipStream_t stream) {
  const float* feature    = (const float*)d_in[0];
  const float* frame_feat = (const float*)d_in[1];
  const float* segments   = (const float*)d_in[2];
  const float* fsegments  = (const float*)d_in[3];
  const float* anchor     = (const float*)d_in[4];
  const float* fanchor    = (const float*)d_in[5];
  const float* W_cur  = (const float*)d_in[7];
  const float* b_cur  = (const float*)d_in[8];
  const float* g_cur  = (const float*)d_in[9];
  const float* be_cur = (const float*)d_in[10];
  const float* W_lr   = (const float*)d_in[11];
  const float* b_lr   = (const float*)d_in[12];
  const float* g_lr   = (const float*)d_in[13];
  const float* be_lr  = (const float*)d_in[14];
  const float* W_roi  = (const float*)d_in[15];
  const float* b_roi  = (const float*)d_in[16];
  const float* g_roi  = (const float*)d_in[17];
  const float* be_roi = (const float*)d_in[18];
  const float* W_prop  = (const float*)d_in[19];
  const float* b_prop  = (const float*)d_in[20];
  const float* g_prop  = (const float*)d_in[21];
  const float* be_prop = (const float*)d_in[22];
  const float* W_cp0 = (const float*)d_in[23];
  const float* b_cp0 = (const float*)d_in[24];
  const float* W_cp1 = (const float*)d_in[25];
  const float* b_cp1 = (const float*)d_in[26];
  // KERNEL[order+1] == 16 for every legal order -> K1 is compile-time 16.

  float* ws = (float*)d_ws;
  const size_t o_cur  = 0;                                    // [B,512,128]
  const size_t o_lr   = o_cur  + (size_t)BBc * PCc * TTc;     // [B,1024,128]
  const size_t o_cat  = o_lr   + (size_t)BBc * 2 * PCc * TTc; // [B,3072,128]
  const size_t o_pool = o_cat  + (size_t)BBc * 6 * PCc * TTc; // [B,512,128]
  const size_t o_fc   = o_pool + (size_t)BBc * PCc * TTc;     // [B,512,128]
  const size_t o_tmp  = o_fc   + (size_t)BBc * PCc * TTc;     // [B,512,128]
  const size_t o_st   = o_tmp  + (size_t)BBc * PCc * TTc;
  const size_t o_idx  = o_st + 256;
  const size_t o_wp   = o_idx + 12288 + 64;       // packed weights, reused (max 4.19M dwords)
  const size_t o_xp   = o_wp + (size_t)PCc * 16384 / 2; // packed activations, reused (max 2.1M)
  float* cat   = ws + o_cat;
  float* stats = ws + o_st;
  int* idx1 = (int*)(ws + o_idx);               // [B,128,16]
  int* idx0 = idx1 + BBc * TTc * 16;            // [B,128,32]
  unsigned int* wp = (unsigned int*)(ws + o_wp);
  unsigned int* xp = (unsigned int*)(ws + o_xp);

  float* out0   = (float*)d_out;                   // [B,512,128]
  float* featlr = out0 + (size_t)BBc * CINc * TTc; // [B,1024,128]

  const long ldcat = (long)6 * PCc * TTc;
  const dim3 gM512(2, PCc / 128, BBc);
  const dim3 gM1024(2, 2 * PCc / 128, BBc);
  const dim3 gStats(GRPS, BBc);
  const int KBIG = 16384;
  const int nPK512  = BBc * PCc * TTc;       // 131072 elements / raw tensors
  auto blocks = [](int n) { return (n + 255) / 256; };

  k_mkidx<<<blocks(BBc*TTc*16), 256, 0, stream>>>(anchor,  idx1, TTc, 16, BBc*TTc*16);
  k_mkidx<<<blocks(BBc*TTc*32), 256, 0, stream>>>(fanchor, idx0, T2c, 32, BBc*TTc*32);

  // pack feature once; reused for both cur and lr GEMMs
  k_pack_b<<<blocks(BBc*CINc*TTc/2), 256, 0, stream>>>(feature, xp, CINc, BBc*CINc*TTc/2);

  // fm_short -> cat[1536:2048]
  k_pack_a<<<blocks(PCc*CINc/2), 256, 0, stream>>>(W_cur, wp, CINc, PCc*CINc/2);
  k_gemm_pk<<<gM512, 256, 0, stream>>>(wp, b_cur, xp, ws + o_cur, CINc/32, (long)PCc*TTc, 0);
  k_gn_stats<<<gStats, 256, 0, stream>>>(ws + o_cur, stats, PCc, TTc);
  k_gn_apply<<<blocks(nPK512), 256, 0, stream>>>(ws + o_cur, stats, g_cur, be_cur,
                                                 cat, PCc, TTc, 1536, 6*PCc, nPK512);
  // feat_lr -> second output region
  k_pack_a<<<blocks(2*PCc*CINc/2), 256, 0, stream>>>(W_lr, wp, CINc, 2*PCc*CINc/2);
  k_gemm_pk<<<gM1024, 256, 0, stream>>>(wp, b_lr, xp, ws + o_lr, CINc/32, (long)2*PCc*TTc, 0);
  k_gn_stats<<<gStats, 256, 0, stream>>>(ws + o_lr, stats, 2*PCc, TTc);
  k_gn_apply<<<blocks(2*nPK512), 256, 0, stream>>>(ws + o_lr, stats, g_lr, be_lr,
                                                   featlr, 2*PCc, TTc, 0, 2*PCc, 2*nPK512);
  // prop_feature -> cat[512:1536]
  k_bmax<<<blocks(2*nPK512), 256, 0, stream>>>(featlr, segments, cat, 2*PCc, TTc, 512, 6*PCc, 2*nPK512);
  // pooled frame features -> roi unit -> cat[0:512]
  k_bmax<<<blocks(nPK512), 256, 0, stream>>>(frame_feat, fsegments, ws + o_pool, PCc, T2c, 0, PCc, nPK512);
  k_pack_b<<<blocks(nPK512/2), 256, 0, stream>>>(ws + o_pool, xp, PCc, nPK512/2);
  k_pack_a<<<blocks(PCc*PCc/2), 256, 0, stream>>>(W_roi, wp, PCc, PCc*PCc/2);
  k_gemm_pk<<<gM512, 256, 0, stream>>>(wp, b_roi, xp, ws + o_tmp, PCc/32, (long)PCc*TTc, 0);
  k_gn_stats<<<gStats, 256, 0, stream>>>(ws + o_tmp, stats, PCc, TTc);
  k_gn_apply<<<blocks(nPK512), 256, 0, stream>>>(ws + o_tmp, stats, g_roi, be_roi,
                                                 cat, PCc, TTc, 0, 6*PCc, nPK512);
  // ConFeature: gather(feat_lr) packed + GEMM (K=16384) -> cat[2048:2560]
  k_gather_pack<<<blocks(BBc*KBIG*TTc/2), 256, 0, stream>>>(featlr, idx1, xp, KBIG, 4, TTc,
                                                            2*PCc, BBc*KBIG*TTc/2);
  k_pack_a<<<blocks(PCc*KBIG/2), 256, 0, stream>>>(W_cp1, wp, KBIG, PCc*KBIG/2);
  k_gemm_pk<<<gM512, 256, 0, stream>>>(wp, b_cp1, xp, cat, KBIG/32, ldcat, 2048);
  // Frame_con: gather(frame) packed + GEMM, then roi unit -> cat[2560:3072]
  k_gather_pack<<<blocks(BBc*KBIG*TTc/2), 256, 0, stream>>>(frame_feat, idx0, xp, KBIG, 5, T2c,
                                                            PCc, BBc*KBIG*TTc/2);
  k_pack_a<<<blocks(PCc*KBIG/2), 256, 0, stream>>>(W_cp0, wp, KBIG, PCc*KBIG/2);
  k_gemm_pk<<<gM512, 256, 0, stream>>>(wp, b_cp0, xp, ws + o_fc, KBIG/32, (long)PCc*TTc, 0);
  k_pack_b<<<blocks(nPK512/2), 256, 0, stream>>>(ws + o_fc, xp, PCc, nPK512/2);
  k_pack_a<<<blocks(PCc*PCc/2), 256, 0, stream>>>(W_roi, wp, PCc, PCc*PCc/2);
  k_gemm_pk<<<gM512, 256, 0, stream>>>(wp, b_roi, xp, ws + o_tmp, PCc/32, (long)PCc*TTc, 0);
  k_gn_stats<<<gStats, 256, 0, stream>>>(ws + o_tmp, stats, PCc, TTc);
  k_gn_apply<<<blocks(nPK512), 256, 0, stream>>>(ws + o_tmp, stats, g_roi, be_roi,
                                                 cat, PCc, TTc, 2560, 6*PCc, nPK512);
  // out = unit(cat, W_prop)  (K = 3072)
  k_pack_b<<<blocks(BBc*6*PCc*TTc/2), 256, 0, stream>>>(cat, xp, 6*PCc, BBc*6*PCc*TTc/2);
  k_pack_a<<<blocks(CINc*6*PCc/2), 256, 0, stream>>>(W_prop, wp, 6*PCc, CINc*6*PCc/2);
  k_gemm_pk<<<gM512, 256, 0, stream>>>(wp, b_prop, xp, ws + o_tmp, 6*PCc/32, (long)PCc*TTc, 0);
  k_gn_stats<<<gStats, 256, 0, stream>>>(ws + o_tmp, stats, PCc, TTc);
  k_gn_apply<<<blocks(nPK512), 256, 0, stream>>>(ws + o_tmp, stats, g_prop, be_prop,
                                                 out0, PCc, TTc, 0, PCc, nPK512);
}
